// SelectNodeNetwork_87875030876928
// MI455X (gfx1250) — matrix operand order repaired and checked by reading, not compiled, not run on hardware
//
#include <hip/hip_runtime.h>

typedef _Float16 v16h __attribute__((ext_vector_type(16)));
typedef float    v8f  __attribute__((ext_vector_type(8)));

// Problem constants (match reference)
constexpr int kB  = 8;
constexpr int kN  = 8192;
constexpr int kE  = 65536;
constexpr int kBN = kB * kN;           // 65536 node rows

// ---------------- workspace layout (float offsets) ----------------
constexpr size_t O_H0   = 0;                         // (BN,64)
constexpr size_t O_AGG  = O_H0   + (size_t)kBN * 64; // (BN,64)  scatter accumulator
constexpr size_t O_CNT  = O_AGG  + (size_t)kBN * 64; // (BN)     in-degree as float
constexpr size_t O_GP   = O_CNT  + (size_t)kBN;      // (BN,192) agg @ Wc + cnt*bc
constexpr size_t O_UH   = O_GP   + (size_t)kBN * 192;// (BN,128) h @ U[:, :128]
constexpr size_t O_Z    = O_UH   + (size_t)kBN * 128;// (BN,64)
constexpr size_t O_RH   = O_Z    + (size_t)kBN * 64; // (BN,64)
constexpr size_t O_HN   = O_RH   + (size_t)kBN * 64; // (BN,64)  new hidden
constexpr size_t O_WC   = O_HN   + (size_t)kBN * 64; // 64*192   fused W_msg@W_gru
constexpr size_t O_BC   = O_WC   + 64 * 192;         // 192      b_msg@W_gru
constexpr size_t O_POOL = O_BC   + 192;              // B*64
// reused regions
constexpr size_t O_CT = O_AGG;  // (BN,64)  rh @ U[:,128:]   (agg dead by then)
constexpr size_t O_X1 = O_UH;   // (BN,128) fc1 out          (uh dead by then)
constexpr size_t O_X2 = O_GP;   // (BN,64)  fc2 out          (gp dead by then)

__device__ __forceinline__ float sigmoidf_(float x) { return 1.0f / (1.0f + __expf(-x)); }

// ---------------- h0 = concat(annotations, state[:, 32:]) ----------------
__global__ __launch_bounds__(256) void prep_h0_kernel(const float* __restrict__ state,
                                                      const float* __restrict__ ann,
                                                      float* __restrict__ h0) {
  const int tid = blockIdx.x * 256 + threadIdx.x;   // BN*64 threads
  const int row = tid >> 6, d = tid & 63;
  h0[tid] = (d < 32) ? ann[row * 32 + d] : state[tid];
}

// ---------------- Wc = W_msg @ W_gru (64x192); bc = b_msg @ W_gru ----------------
__global__ void wc_kernel(const float* __restrict__ Wmsg, const float* __restrict__ bmsg,
                          const float* __restrict__ Wgru, float* __restrict__ Wc,
                          float* __restrict__ bc) {
  const int tid = blockIdx.x * blockDim.x + threadIdx.x;
  if (tid >= 65 * 192) return;
  const int k = tid / 192, j = tid % 192;
  float s = 0.0f;
  if (k < 64) {
    for (int i = 0; i < 64; ++i) s += Wmsg[k * 64 + i] * Wgru[i * 192 + j];
    Wc[k * 192 + j] = s;
  } else {
    for (int i = 0; i < 64; ++i) s += bmsg[i] * Wgru[i * 192 + j];
    bc[j] = s;
  }
}

// ---------------- edge scatter: agg[tgt] += h0[src]; cnt[tgt] += 1 ----------------
__global__ __launch_bounds__(256) void scatter_kernel(const float* __restrict__ h0,
                                                      const long long* __restrict__ adj,
                                                      float* __restrict__ agg,
                                                      float* __restrict__ cnt) {
  const size_t tid = (size_t)blockIdx.x * 256 + threadIdx.x;  // B*E*16 threads
  const size_t eg  = tid >> 4;                                // global edge id
  const int    q   = (int)(tid & 15);                         // 1/16 row slice (4 floats)
  if (eg >= (size_t)kB * kE) return;
  const int b   = (int)(eg / kE);
  const int src = (int)adj[eg * 2 + 0];
  const int tgt = (int)adj[eg * 2 + 1];
  const float4 v = *(const float4*)(h0 + ((size_t)b * kN + src) * 64 + q * 4);
  float* dst = agg + ((size_t)b * kN + tgt) * 64 + q * 4;
  atomicAdd(dst + 0, v.x);
  atomicAdd(dst + 1, v.y);
  atomicAdd(dst + 2, v.z);
  atomicAdd(dst + 3, v.w);
  if (q == 0) atomicAdd(&cnt[(size_t)b * kN + tgt], 1.0f);
}

// ---------------- generic WMMA GEMM: out = epi(A[BN,K] @ W[K,NOUT]) ----------------
// A: f32 row-major; W: f32 row-major, leading dim LDW; f32->f16 cvt on fragment load.
// Fully static shapes: branchless epilogue, immediate-offset stores.
template <int K, int NOUT, int LDW, bool HAS_BIAS, bool HAS_ROW, bool RELU>
__global__ __launch_bounds__(256) void wmma_gemm_kernel(
    const float* __restrict__ A, const float* __restrict__ W,
    float* __restrict__ out,
    const float* __restrict__ bias, const float* __restrict__ rowvec,
    const float* __restrict__ roww) {
  constexpr int KT = K / 32;   // k-steps of 32
  constexpr int MT = 8;        // m-tiles per wave (weights reused from registers)
  const int wave = threadIdx.x >> 5;
  const int lane = threadIdx.x & 31;
  const int hi   = lane >> 4;                    // lane half (ISA wave32 frag layout)
  const int col  = (lane & 15) + blockIdx.y * 16;

  // B fragments: 32x16 tile of W per k-step; VGPR v<4 -> K=2v, v>=4 -> K=16+2(v-4) (+8*hi)
  v16h bfrag[KT];
#pragma unroll
  for (int t = 0; t < KT; ++t) {
#pragma unroll
    for (int v = 0; v < 8; ++v) {
      const int k = t * 32 + ((v < 4) ? 2 * v : 16 + 2 * (v - 4)) + hi * 8;
      bfrag[t][2 * v]     = (_Float16)W[k * LDW + col];
      bfrag[t][2 * v + 1] = (_Float16)W[(k + 1) * LDW + col];
    }
  }

  float bcol = 0.0f, wcol = 0.0f;
  if constexpr (HAS_BIAS) bcol = bias[col];
  if constexpr (HAS_ROW)  wcol = roww[col];
  const int m      = lane & 15;
  const int mtile0 = (blockIdx.x * 8 + wave) * MT;
  const float* arow0 = A + (size_t)(mtile0 * 16 + m) * K + hi * 8;

#pragma unroll
  for (int it = 0; it < MT; ++it) {
    const int mtile = mtile0 + it;
    const float* arow = arow0 + (size_t)it * 16 * K;
    if (it + 1 < MT) __builtin_prefetch(arow + 16 * K, 0, 0);  // next tile's A row
    v8f acc = {};
#pragma unroll
    for (int t = 0; t < KT; ++t) {
      v16h afrag;
#pragma unroll
      for (int v = 0; v < 8; ++v) {   // A: VGPR v<4 -> K=2v, v>=4 -> K=16+2(v-4) (+8*hi)
        const int k = t * 32 + ((v < 4) ? 2 * v : 16 + 2 * (v - 4));
        const float2 f2 = *(const float2*)(arow + k);
        afrag[2 * v]     = (_Float16)f2.x;
        afrag[2 * v + 1] = (_Float16)f2.y;
      }
      acc = __builtin_amdgcn_wmma_f32_16x16x32_f16(false, afrag, false, bfrag[t],
                                                   (short)0, acc, false, false);
    }
    // epilogue: lane's 8 output rows are rowbase..rowbase+7 (contiguous)
    const int rowbase = mtile * 16 + hi * 8;
    float rv[8];
    if constexpr (HAS_ROW) {
      const float4 r0 = *(const float4*)(rowvec + rowbase);
      const float4 r1 = *(const float4*)(rowvec + rowbase + 4);
      rv[0] = r0.x; rv[1] = r0.y; rv[2] = r0.z; rv[3] = r0.w;
      rv[4] = r1.x; rv[5] = r1.y; rv[6] = r1.z; rv[7] = r1.w;
    }
    float* outp = out + (size_t)rowbase * NOUT + col;
#pragma unroll
    for (int i = 0; i < 8; ++i) {     // D: VGPR i -> row rowbase + i
      float v = acc[i];
      if constexpr (HAS_BIAS) v += bcol;
      if constexpr (HAS_ROW)  v += rv[i] * wcol;
      if constexpr (RELU)     v = fmaxf(v, 0.0f);
      outp[i * NOUT] = v;
    }
  }
}

// ---------------- GRU elementwise: z, r, rh ----------------
__global__ __launch_bounds__(256) void ew_zr_kernel(const float* __restrict__ gp,
                                                    const float* __restrict__ uh,
                                                    const float* __restrict__ bgru,
                                                    const float* __restrict__ h0,
                                                    float* __restrict__ zbuf,
                                                    float* __restrict__ rh) {
  const int tid = blockIdx.x * 256 + threadIdx.x;   // BN*64
  const int row = tid >> 6, d = tid & 63;
  const float z = sigmoidf_(gp[(size_t)row * 192 + d]      + uh[(size_t)row * 128 + d]      + bgru[d]);
  const float r = sigmoidf_(gp[(size_t)row * 192 + 64 + d] + uh[(size_t)row * 128 + 64 + d] + bgru[64 + d]);
  zbuf[tid] = z;
  rh[tid]   = r * h0[tid];
}

// ---------------- GRU elementwise: c, h_new ----------------
__global__ __launch_bounds__(256) void ew_c_kernel(const float* __restrict__ gp,
                                                   const float* __restrict__ ct,
                                                   const float* __restrict__ bgru,
                                                   const float* __restrict__ zbuf,
                                                   const float* __restrict__ h0,
                                                   float* __restrict__ hnew) {
  const int tid = blockIdx.x * 256 + threadIdx.x;   // BN*64
  const int row = tid >> 6, d = tid & 63;
  const float c = tanhf(gp[(size_t)row * 192 + 128 + d] + ct[tid] + bgru[128 + d]);
  const float z = zbuf[tid];
  hnew[tid] = (1.0f - z) * h0[tid] + z * c;
}

// ---------------- attention pool (coalesced): pool[b,f] += sum_n x2[b,n,f]*attn_w[n] ----
// grid = (B, kN/256); block covers a 256-row slab: f = t&63 across a row, g = t>>6 row group
__global__ __launch_bounds__(256) void pool_kernel(const float* __restrict__ x2,
                                                   const float* __restrict__ attn_w,
                                                   float* __restrict__ pool) {
  __shared__ float red[256];
  const int b = blockIdx.x;
  const int n0 = blockIdx.y * 256;
  const int f = threadIdx.x & 63, g = threadIdx.x >> 6;   // g in 0..3
  float s = 0.0f;
  for (int n = n0 + g; n < n0 + 256; n += 4)
    s += x2[((size_t)b * kN + n) * 64 + f] * attn_w[n];
  red[threadIdx.x] = s;
  __syncthreads();
  if (g == 0)
    atomicAdd(&pool[b * 64 + f], red[f] + red[64 + f] + red[128 + f] + red[192 + f]);
}

// ---------------- fc3 + softmax + mask, one block per batch ----------------
__global__ __launch_bounds__(1024) void final_kernel(const float* __restrict__ pool,
                                                     const float* __restrict__ attn_b,
                                                     const float* __restrict__ fc3w,
                                                     const float* __restrict__ fc3b,
                                                     const int* __restrict__ mask,
                                                     float* __restrict__ out) {
  __shared__ float ps[64];
  __shared__ float red[1024];
  const int b = blockIdx.x, t = threadIdx.x;
  if (t < 64) ps[t] = pool[b * 64 + t] + attn_b[0];
  __syncthreads();
  float l[8];
  float mx = -3.0e38f;
#pragma unroll
  for (int j = 0; j < 8; ++j) {
    const int n = t + j * 1024;
    float s = fc3b[n];
    for (int f = 0; f < 64; ++f) s += ps[f] * fc3w[(size_t)f * kN + n];
    l[j] = s;
    mx = fmaxf(mx, s);
  }
  red[t] = mx;
  __syncthreads();
  for (int st = 512; st > 0; st >>= 1) {
    if (t < st) red[t] = fmaxf(red[t], red[t + st]);
    __syncthreads();
  }
  mx = red[0];
  __syncthreads();
  float se = 0.0f;
#pragma unroll
  for (int j = 0; j < 8; ++j) { l[j] = __expf(l[j] - mx); se += l[j]; }
  red[t] = se;
  __syncthreads();
  for (int st = 512; st > 0; st >>= 1) {
    if (t < st) red[t] += red[t + st];
    __syncthreads();
  }
  const float inv = 1.0f / red[0];
#pragma unroll
  for (int j = 0; j < 8; ++j) {
    const int n = t + j * 1024;
    out[(size_t)b * kN + n] = (mask[(size_t)b * kN + n] > 0) ? l[j] * inv : -1000000.0f;
  }
}

// ---------------- host launcher ----------------
extern "C" void kernel_launch(void* const* d_in, const int* in_sizes, int n_in,
                              void* d_out, int out_size, void* d_ws, size_t ws_size,
                              hipStream_t stream) {
  const float*     state = (const float*)d_in[0];
  const float*     ann   = (const float*)d_in[1];
  const float*     spill = (const float*)d_in[2];
  const float*     Wmsg  = (const float*)d_in[3];
  const float*     bmsg  = (const float*)d_in[4];
  const float*     Wgru  = (const float*)d_in[5];
  const float*     Ugru  = (const float*)d_in[6];
  const float*     bgru  = (const float*)d_in[7];
  const float*     fc1w  = (const float*)d_in[8];
  const float*     fc1b  = (const float*)d_in[9];
  const float*     fc2w  = (const float*)d_in[10];
  const float*     fc2b  = (const float*)d_in[11];
  const float*     attnw = (const float*)d_in[12];
  const float*     attnb = (const float*)d_in[13];
  const float*     fc3w  = (const float*)d_in[14];
  const float*     fc3b  = (const float*)d_in[15];
  const long long* adj   = (const long long*)d_in[16];
  const int*       amask = (const int*)d_in[17];
  float* outp = (float*)d_out;

  float* ws   = (float*)d_ws;
  float* h0   = ws + O_H0;
  float* agg  = ws + O_AGG;
  float* cnt  = ws + O_CNT;
  float* gp   = ws + O_GP;
  float* uh   = ws + O_UH;
  float* zbuf = ws + O_Z;
  float* rh   = ws + O_RH;
  float* hnew = ws + O_HN;
  float* Wc   = ws + O_WC;
  float* bc   = ws + O_BC;
  float* pool = ws + O_POOL;
  float* ctb  = ws + O_CT;
  float* x1   = ws + O_X1;
  float* x2   = ws + O_X2;

  // h0 build + fused weight precompute
  prep_h0_kernel<<<kBN * 64 / 256, 256, 0, stream>>>(state, ann, h0);
  wc_kernel<<<(65 * 192 + 255) / 256, 256, 0, stream>>>(Wmsg, bmsg, Wgru, Wc, bc);

  // zero scatter accumulator + counts (contiguous) and pool, then edge scatter
  hipMemsetAsync(agg, 0, (size_t)(kBN * 64 + kBN) * sizeof(float), stream);
  hipMemsetAsync(pool, 0, (size_t)(kB * 64) * sizeof(float), stream);
  scatter_kernel<<<(kB * kE * 16) / 256, 256, 0, stream>>>(h0, adj, agg, cnt);

  // gp = agg @ Wc + cnt*bc            (BN,192)
  wmma_gemm_kernel<64, 192, 192, false, true, false>
      <<<dim3(64, 12), 256, 0, stream>>>(agg, Wc, gp, nullptr, cnt, bc);
  // uh = h0 @ U_gru[:, :128]          (BN,128)
  wmma_gemm_kernel<64, 128, 192, false, false, false>
      <<<dim3(64, 8), 256, 0, stream>>>(h0, Ugru, uh, nullptr, nullptr, nullptr);
  // z, r, rh
  ew_zr_kernel<<<kBN * 64 / 256, 256, 0, stream>>>(gp, uh, bgru, h0, zbuf, rh);
  // ct = rh @ U_gru[:, 128:]          (BN,64)
  wmma_gemm_kernel<64, 64, 192, false, false, false>
      <<<dim3(64, 4), 256, 0, stream>>>(rh, Ugru + 128, ctb, nullptr, nullptr, nullptr);
  // c, h_new
  ew_c_kernel<<<kBN * 64 / 256, 256, 0, stream>>>(gp, ctb, bgru, zbuf, h0, hnew);
  // x1 = relu(h_new @ fc1_w + fc1_b)  (BN,128)
  wmma_gemm_kernel<64, 128, 128, true, false, true>
      <<<dim3(64, 8), 256, 0, stream>>>(hnew, fc1w, x1, fc1b, nullptr, nullptr);
  // x2 = relu(x1 @ fc2_w[1:] + spill*fc2_w[0] + fc2_b)   (BN,64)
  wmma_gemm_kernel<128, 64, 64, true, true, true>
      <<<dim3(64, 4), 256, 0, stream>>>(x1, fc2w + 64, x2, fc2b, spill, fc2w);
  // attention pool + fc3 + softmax + mask
  pool_kernel<<<dim3(kB, kN / 256), 256, 0, stream>>>(x2, attnw, pool);
  final_kernel<<<kB, 1024, 0, stream>>>(pool, attnb, fc3w, fc3b, amask, outp);
}
